// graph_71906342470140
// MI455X (gfx1250) — compile-verified, matrix operand order checked by reading, not run on hardware
//
#include <hip/hip_runtime.h>
#include <hip/hip_bf16.h>

typedef unsigned short u16;
typedef __bf16 v16bf __attribute__((ext_vector_type(16)));
typedef __bf16 v8bf  __attribute__((ext_vector_type(8)));
typedef float  v8f   __attribute__((ext_vector_type(8)));
typedef int    v4i   __attribute__((ext_vector_type(4)));

union Frag { v16bf v; v8bf h[2]; };

__device__ __forceinline__ u16 f2bf(float f) {
    unsigned u = __float_as_uint(f);
    unsigned r = u + 0x7fffu + ((u >> 16) & 1u);
    return (u16)(r >> 16);
}
__device__ __forceinline__ float bf2f(u16 h) {
    return __uint_as_float(((unsigned)h) << 16);
}
__device__ __forceinline__ float ld_f(const float* p, size_t i) { return p[i]; }
__device__ __forceinline__ float ld_f(const u16*   p, size_t i) { return bf2f(p[i]); }

// ---- CDNA5 async global->LDS copy (direct path, ASYNCcnt-tracked), with
// ---- portable fallback to VGPR round-trip if the builtins are absent.
#if defined(__HIP_DEVICE_COMPILE__) &&                                         \
    __has_builtin(__builtin_amdgcn_global_load_async_to_lds_b128) &&           \
    __has_builtin(__builtin_amdgcn_s_wait_asynccnt)
#define HAVE_ASYNC_LDS 1
#else
#define HAVE_ASYNC_LDS 0
#endif

#if HAVE_ASYNC_LDS
typedef __attribute__((address_space(1))) v4i* gv4i_p;   // global int4*
typedef __attribute__((address_space(3))) v4i* lv4i_p;   // LDS int4*
#endif

__device__ __forceinline__ void async_cp32B(const u16* gsrc, u16* ldst) {
#if HAVE_ASYNC_LDS
    __builtin_amdgcn_global_load_async_to_lds_b128(
        (gv4i_p)(v4i*)gsrc, (lv4i_p)(v4i*)ldst, 0, 0);
    __builtin_amdgcn_global_load_async_to_lds_b128(
        (gv4i_p)(v4i*)(gsrc + 8), (lv4i_p)(v4i*)(ldst + 8), 0, 0);
#else
    const uint4* s = reinterpret_cast<const uint4*>(gsrc);
    uint4* d = reinterpret_cast<uint4*>(ldst);
    d[0] = s[0];
    d[1] = s[1];
#endif
}
__device__ __forceinline__ void wait_async_lds() {
#if HAVE_ASYNC_LDS
    __builtin_amdgcn_s_wait_asynccnt(0);
#endif
}

// ---------------------------------------------------------------------------
// Generic bf16 WMMA GEMM:  C[M,N] = A[M,K] * B^T   where B is stored [N][K]
// (N-major so the B fragment loader matches the A loader exactly).
// Tile: 128x128, K-step 32, double-buffered LDS + async global->LDS staging.
// 256 threads = 8 waves; each wave owns 64(M) x 32(N) -> 4x2 subtiles of
// v_wmma_f32_16x16x32_bf16 (f32 accumulate).
// EPI: 0 f32 store | 1 bf16 store | 2 BN+ReLU bf16 | 3 bias+ReLU bf16
//      4 (+residual bf16) f32 store
// CONV3A: A generated as implicit-im2col of NHWC [4,64,64,1024], K = 9*1024
//         (tap-major), zero-fill at borders (direct LDS store for zeros).
// ---------------------------------------------------------------------------
constexpr int KPAD = 40;   // 32 data + 8 pad bf16 per LDS row (80B, 16B-aligned)

template<int EPI, bool CONV3A>
__global__ __launch_bounds__(256)
void gemm_bf16_kernel(const u16* __restrict__ A, int lda,
                      const u16* __restrict__ B, int ldb,
                      void* __restrict__ Cv, int ldc,
                      int M, int N, int K,
                      const float* __restrict__ p0,
                      const float* __restrict__ p1,
                      const u16* __restrict__ addsrc)
{
    __shared__ alignas(16) u16 ldsA[2][128 * KPAD];
    __shared__ alignas(16) u16 ldsB[2][128 * KPAD];

    const int t     = threadIdx.x;
    const int lane  = t & 31;
    const int wave  = t >> 5;
    const int waveM = wave >> 2;     // 0..1  -> 64-row group
    const int waveN = wave & 3;      // 0..3  -> 32-col group
    const int m0    = blockIdx.y * 128;
    const int n0    = blockIdx.x * 128;

    v8f acc[4][2];
    #pragma unroll
    for (int mi = 0; mi < 4; ++mi)
        #pragma unroll
        for (int ni = 0; ni < 2; ++ni)
            #pragma unroll
            for (int e = 0; e < 8; ++e) acc[mi][ni][e] = 0.0f;

    // staging assignment: each thread copies 16 contiguous bf16 (2 x 16B)
    const int srow = t >> 1;
    const int colb = (t & 1) * 16;

    const int hf  = lane >> 4;       // which K-half this lane holds
    const int l16 = lane & 15;

    auto stage = [&](int buf, int kt) {
        // ---- A tile [128 x 32] ----
        u16* dstA = &ldsA[buf][srow * KPAD + colb];
        if (CONV3A) {
            const int kg  = kt + colb;      // 16-chunk never crosses a tap
            const int tap = kg >> 10;
            const int ci  = kg & 1023;
            const int ky  = tap / 3, kx = tap % 3;
            const int mg  = m0 + srow;
            const int b   = mg >> 12, hw = mg & 4095;
            const int h   = hw >> 6,  w  = hw & 63;
            const int hs  = h + ky - 1, wsrc = w + kx - 1;
            if ((unsigned)hs < 64u && (unsigned)wsrc < 64u) {
                async_cp32B(A + ((size_t)((b << 12) + (hs << 6) + wsrc) << 10) + ci,
                            dstA);
            } else {
                uint4* d = reinterpret_cast<uint4*>(dstA);
                d[0] = make_uint4(0u, 0u, 0u, 0u);
                d[1] = make_uint4(0u, 0u, 0u, 0u);
            }
        } else {
            async_cp32B(A + (size_t)(m0 + srow) * lda + kt + colb, dstA);
        }
        // ---- B tile [128 x 32] (B stored [N][K]) ----
        async_cp32B(B + (size_t)(n0 + srow) * ldb + kt + colb,
                    &ldsB[buf][srow * KPAD + colb]);
    };

    // prologue: fill buffer 0
    stage(0, 0);
    wait_async_lds();
    __syncthreads();

    int buf = 0;
    for (int kt = 0; kt < K; kt += 32) {
        // overlap: start next tile's async copy into the other buffer
        if (kt + 32 < K) stage(buf ^ 1, kt + 32);

        // ---- fragments + WMMA from current buffer ----
        Frag af[4], bfv[2];
        #pragma unroll
        for (int mi = 0; mi < 4; ++mi) {
            const int r = waveM * 64 + mi * 16 + l16;
            af[mi].h[0] = *reinterpret_cast<const v8bf*>(&ldsA[buf][r * KPAD + hf * 8]);
            af[mi].h[1] = *reinterpret_cast<const v8bf*>(&ldsA[buf][r * KPAD + 16 + hf * 8]);
        }
        #pragma unroll
        for (int ni = 0; ni < 2; ++ni) {
            const int r = waveN * 32 + ni * 16 + l16;
            bfv[ni].h[0] = *reinterpret_cast<const v8bf*>(&ldsB[buf][r * KPAD + hf * 8]);
            bfv[ni].h[1] = *reinterpret_cast<const v8bf*>(&ldsB[buf][r * KPAD + 16 + hf * 8]);
        }
        #pragma unroll
        for (int mi = 0; mi < 4; ++mi)
            #pragma unroll
            for (int ni = 0; ni < 2; ++ni)
                acc[mi][ni] = __builtin_amdgcn_wmma_f32_16x16x32_bf16(
                    false, af[mi].v, false, bfv[ni].v, (short)0, acc[mi][ni],
                    false, false);

        // our async writes to buf^1 must land, then all waves rendezvous
        wait_async_lds();
        __syncthreads();
        buf ^= 1;
    }

    // ---- epilogue ----
    #pragma unroll
    for (int mi = 0; mi < 4; ++mi) {
        #pragma unroll
        for (int ni = 0; ni < 2; ++ni) {
            #pragma unroll
            for (int r = 0; r < 8; ++r) {
                const int m = m0 + waveM * 64 + mi * 16 + hf * 8 + r;
                const int n = n0 + waveN * 32 + ni * 16 + l16;
                float c = acc[mi][ni][r];
                if (EPI == 0) {
                    ((float*)Cv)[(size_t)m * ldc + n] = c;
                } else if (EPI == 1) {
                    ((u16*)Cv)[(size_t)m * ldc + n] = f2bf(c);
                } else if (EPI == 2) {
                    float v = c * p0[n] + p1[n];
                    v = v > 0.0f ? v : 0.0f;
                    ((u16*)Cv)[(size_t)m * ldc + n] = f2bf(v);
                } else if (EPI == 3) {
                    float v = c + p0[n];
                    v = v > 0.0f ? v : 0.0f;
                    ((u16*)Cv)[(size_t)m * ldc + n] = f2bf(v);
                } else { // EPI == 4
                    float v = c + bf2f(addsrc[(size_t)m * ldc + n]);
                    ((float*)Cv)[(size_t)m * ldc + n] = v;
                }
            }
        }
    }
}

// ---------------------------------------------------------------------------
// Elementwise / prep kernels
// ---------------------------------------------------------------------------

// concat(T1,T2) NCHW f32 -> Xc NHWC bf16  [4][4096][1024]
__global__ __launch_bounds__(256)
void concat_cast_kernel(const float* __restrict__ T1, const float* __restrict__ T2,
                        u16* __restrict__ Xc)
{
    const int idx = blockIdx.x * 256 + threadIdx.x;   // < 4*4096*1024
    const int c  = idx & 1023;
    const int hw = (idx >> 10) & 4095;
    const int b  = idx >> 22;
    float v;
    if (c < 512) v = T1[((size_t)(b * 512 + c)) * 4096 + hw];
    else         v = T2[((size_t)(b * 512 + (c - 512))) * 4096 + hw];
    Xc[idx] = f2bf(v);
}

// w_conv3 [O=1024][I=1024][3][3] f32 -> wB3 bf16 [n=O][k=(ky*3+kx)*1024+ci]
__global__ __launch_bounds__(256)
void prep_w3_kernel(const float* __restrict__ w3, u16* __restrict__ wB3)
{
    const int idx = blockIdx.x * 256 + threadIdx.x;   // < 1024*9216
    const int n   = idx / 9216;
    const int r   = idx - n * 9216;
    const int tap = r >> 10;
    const int ci  = r & 1023;
    const int ky  = tap / 3, kx = tap % 3;
    wB3[idx] = f2bf(w3[(((size_t)n * 1024 + ci) * 3 + ky) * 3 + kx]);
}

// w_conv1 [512][1024] cast + BN scale/shift precompute
__global__ __launch_bounds__(256)
void prep_w1_bn_kernel(const float* __restrict__ w1,
                       const float* __restrict__ g, const float* __restrict__ be,
                       const float* __restrict__ mu, const float* __restrict__ var,
                       u16* __restrict__ wB1, float* __restrict__ bns,
                       float* __restrict__ bnsh)
{
    const int idx = blockIdx.x * 256 + threadIdx.x;   // < 512*1024
    wB1[idx] = f2bf(w1[idx]);
    if (idx < 512) {
        float s = g[idx] * rsqrtf(var[idx] + 1e-5f);
        bns[idx]  = s;
        bnsh[idx] = be[idx] - mu[idx] * s;
    }
}

__global__ __launch_bounds__(256)
void cast_lin_kernel(const float* __restrict__ lw, u16* __restrict__ lwb)
{
    const int idx = blockIdx.x * 256 + threadIdx.x;   // < 512*512
    lwb[idx] = f2bf(lw[idx]);
}

// Row softmax over 4096, f32 in -> bf16 out
__global__ __launch_bounds__(256)
void softmax_kernel(const float* __restrict__ S, u16* __restrict__ P)
{
    __shared__ float red[256];
    const int row = blockIdx.x;
    const int t   = threadIdx.x;
    const float* srow = S + (size_t)row * 4096;
    float vals[16];
    float mx = -1e30f;
    #pragma unroll
    for (int j = 0; j < 16; ++j) { vals[j] = srow[t + j * 256]; mx = fmaxf(mx, vals[j]); }
    red[t] = mx; __syncthreads();
    for (int s = 128; s > 0; s >>= 1) { if (t < s) red[t] = fmaxf(red[t], red[t + s]); __syncthreads(); }
    mx = red[0]; __syncthreads();
    float sum = 0.0f;
    #pragma unroll
    for (int j = 0; j < 16; ++j) { vals[j] = __expf(vals[j] - mx); sum += vals[j]; }
    red[t] = sum; __syncthreads();
    for (int s = 128; s > 0; s >>= 1) { if (t < s) red[t] += red[t + s]; __syncthreads(); }
    const float inv = 1.0f / red[0];
    #pragma unroll
    for (int j = 0; j < 16; ++j) P[(size_t)row * 4096 + t + j * 256] = f2bf(vals[j] * inv);
}

// LayerNorm over D=512; one wave per row; 8 rows per block.
// MODE 0: bf16 out [row][c].  MODE 1: f32 out NCHW [b][c][hw] (final output).
template<typename TIN, int MODE>
__global__ __launch_bounds__(256)
void layernorm_kernel(const TIN* __restrict__ X, const float* __restrict__ w,
                      const float* __restrict__ bb, void* __restrict__ outv)
{
    const int lane = threadIdx.x & 31;
    const int wv   = threadIdx.x >> 5;
    const int row  = blockIdx.x * 8 + wv;
    const TIN* xr  = X + (size_t)row * 512;
    float v[16];
    float s = 0.0f;
    #pragma unroll
    for (int j = 0; j < 16; ++j) { v[j] = ld_f(xr, lane + j * 32); s += v[j]; }
    #pragma unroll
    for (int o = 16; o > 0; o >>= 1) s += __shfl_xor(s, o);
    const float mean = s * (1.0f / 512.0f);
    float q = 0.0f;
    #pragma unroll
    for (int j = 0; j < 16; ++j) { float d = v[j] - mean; q += d * d; }
    #pragma unroll
    for (int o = 16; o > 0; o >>= 1) q += __shfl_xor(q, o);
    const float inv = rsqrtf(q * (1.0f / 512.0f) + 1e-12f);
    #pragma unroll
    for (int j = 0; j < 16; ++j) {
        const int c = lane + j * 32;
        const float y = w[c] * (v[j] - mean) * inv + bb[c];
        if (MODE == 0) {
            ((u16*)outv)[(size_t)row * 512 + c] = f2bf(y);
        } else {
            const int b = row >> 12, hw = row & 4095;
            ((float*)outv)[((size_t)b * 512 + c) * 4096 + hw] = y;
        }
    }
}

// bf16 transpose: in [rows][cols] -> out [cols][rows]  (rows=4096, cols=512)
__global__ __launch_bounds__(256)
void transpose_bf16_kernel(const u16* __restrict__ in, u16* __restrict__ out,
                           int rows, int cols)
{
    __shared__ u16 tile[32][33];
    const int tx = threadIdx.x & 31, ty = threadIdx.x >> 5;
    const int bx = blockIdx.x, by = blockIdx.y;
    #pragma unroll
    for (int i = 0; i < 32; i += 8)
        tile[ty + i][tx] = in[(size_t)(by * 32 + ty + i) * cols + bx * 32 + tx];
    __syncthreads();
    #pragma unroll
    for (int i = 0; i < 32; i += 8)
        out[(size_t)(bx * 32 + ty + i) * rows + by * 32 + tx] = tile[tx][ty + i];
}

// ---------------------------------------------------------------------------
// Host orchestration
// ---------------------------------------------------------------------------
extern "C" void kernel_launch(void* const* d_in, const int* in_sizes, int n_in,
                              void* d_out, int out_size, void* d_ws, size_t ws_size,
                              hipStream_t stream)
{
    (void)in_sizes; (void)n_in; (void)out_size; (void)ws_size;
    const float* T1   = (const float*)d_in[0];
    const float* T2   = (const float*)d_in[1];
    const float* w3   = (const float*)d_in[2];
    const float* w1   = (const float*)d_in[3];
    const float* bng  = (const float*)d_in[4];
    const float* bnb  = (const float*)d_in[5];
    const float* bnm  = (const float*)d_in[6];
    const float* bnv  = (const float*)d_in[7];
    const float* ln1w = (const float*)d_in[8];
    const float* ln1b = (const float*)d_in[9];
    const float* linw = (const float*)d_in[10];
    const float* linb = (const float*)d_in[11];
    const float* ln2w = (const float*)d_in[12];
    const float* ln2b = (const float*)d_in[13];
    float* out = (float*)d_out;

    char* base = (char*)d_ws;
    size_t off = 0;
    auto take = [&](size_t bytes) -> void* {
        void* p = base + off;
        off += (bytes + 255) & ~(size_t)255;
        return p;
    };

    u16*   Xc    = (u16*)  take((size_t)16384 * 1024 * 2);  // concat NHWC bf16
    u16*   wB3   = (u16*)  take((size_t)1024 * 9216 * 2);   // conv3 weights [N][K]
    u16*   wB1   = (u16*)  take((size_t)512 * 1024 * 2);    // conv1 weights [N][K]
    u16*   linWb = (u16*)  take((size_t)512 * 512 * 2);     // linear weights [N][K]
    float* bns   = (float*)take(512 * 4);
    float* bnsh  = (float*)take(512 * 4);
    u16*   Y1    = (u16*)  take((size_t)16384 * 1024 * 2);  // conv3 out bf16
    u16*   xb    = (u16*)  take((size_t)16384 * 512 * 2);   // x bf16
    u16*   xT    = (u16*)  take((size_t)512 * 4096 * 2);    // per-batch x^T
    float* sc    = (float*)take((size_t)4096 * 4096 * 4);   // scores f32
    u16*   pr    = (u16*)  take((size_t)4096 * 4096 * 2);   // probs bf16
    float* sbuf  = (float*)take((size_t)16384 * 512 * 4);   // x + rel f32
    u16*   hbuf  = (u16*)  take((size_t)16384 * 512 * 2);   // LN1 out
    u16*   tbuf  = (u16*)  take((size_t)16384 * 512 * 2);   // linear out
    u16*   relb  = (u16*)  take((size_t)16384 * 512 * 2);   // block output

    // --- prep ---
    concat_cast_kernel<<<65536, 256, 0, stream>>>(T1, T2, Xc);
    prep_w3_kernel<<<36864, 256, 0, stream>>>(w3, wB3);
    prep_w1_bn_kernel<<<2048, 256, 0, stream>>>(w1, bng, bnb, bnm, bnv, wB1, bns, bnsh);
    cast_lin_kernel<<<1024, 256, 0, stream>>>(linw, linWb);

    // --- conv3x3 (implicit GEMM, K = 9*1024) -> Y1 bf16 ---
    gemm_bf16_kernel<1, true><<<dim3(8, 128), 256, 0, stream>>>(
        Xc, 0, wB3, 9216, Y1, 1024, 16384, 1024, 9216, nullptr, nullptr, nullptr);

    // --- conv1x1 + BN + ReLU -> x bf16 ---
    gemm_bf16_kernel<2, false><<<dim3(4, 128), 256, 0, stream>>>(
        Y1, 1024, wB1, 1024, xb, 512, 16384, 512, 1024, bns, bnsh, nullptr);

    // --- two attention + reps blocks (shared params) ---
    const u16* xin = xb;
    for (int blk = 0; blk < 2; ++blk) {
        for (int b = 0; b < 4; ++b) {
            const u16* xbat = xin + (size_t)b * 4096 * 512;
            // scores = x x^T  (B stored [N][K] == x row-major)
            gemm_bf16_kernel<0, false><<<dim3(32, 32), 256, 0, stream>>>(
                xbat, 512, xbat, 512, sc, 4096, 4096, 4096, 512,
                nullptr, nullptr, nullptr);
            softmax_kernel<<<4096, 256, 0, stream>>>(sc, pr);
            transpose_bf16_kernel<<<dim3(16, 128), 256, 0, stream>>>(xbat, xT, 4096, 512);
            // rel = attn * x ; fused residual add (+x) -> f32
            gemm_bf16_kernel<4, false><<<dim3(4, 32), 256, 0, stream>>>(
                pr, 4096, xT, 4096, sbuf + (size_t)b * 4096 * 512, 512,
                4096, 512, 4096, nullptr, nullptr, xbat);
        }
        layernorm_kernel<float, 0><<<2048, 256, 0, stream>>>(sbuf, ln1w, ln1b, hbuf);
        // linear (out = h * lin_w^T) + bias + ReLU
        gemm_bf16_kernel<3, false><<<dim3(4, 128), 256, 0, stream>>>(
            hbuf, 512, linWb, 512, tbuf, 512, 16384, 512, 512,
            linb, nullptr, nullptr);
        if (blk == 0)
            layernorm_kernel<u16, 0><<<2048, 256, 0, stream>>>(tbuf, ln2w, ln2b, relb);
        else
            layernorm_kernel<u16, 1><<<2048, 256, 0, stream>>>(tbuf, ln2w, ln2b, out);
        xin = relb;
    }
}
